// ConditionedMambaTransition_30305289240704
// MI455X (gfx1250) — compile-verified, matrix operand order checked by reading, not run on hardware
//
#include <hip/hip_runtime.h>

// ---------------- problem constants ----------------
#define BB    1024
#define TT    64
#define DD    256
#define DS    256
#define DU    64
#define NOBS  80          // NP_*2 + NI
#define HZd   512
#define SIN   577
#define KP1   608         // SIN padded to multiple of 32
#define H1d   512
#define EPSbn 1e-5f
#define NB_   16384       // DD*DU
#define NC_   20480       // NOBS*DD

typedef unsigned short u16;
typedef __attribute__((ext_vector_type(16))) __bf16 v16bf;
typedef __attribute__((ext_vector_type(8)))  float  v8f;

union Frag { v16bf v; uint4 q[2]; };

__device__ __forceinline__ u16 f2bf(float x) {
    unsigned int u = __float_as_uint(x);
    unsigned int r = u + 0x7fffu + ((u >> 16) & 1u);
    return (u16)(r >> 16);
}

// CDNA5 async copy: LDS[lds_off] <= MEM[gaddr] (16B), tracked with ASYNCcnt.
__device__ __forceinline__ void async_b128(unsigned lds_off, const void* gaddr) {
    asm volatile("global_load_async_to_lds_b128 %0, %1, off"
                 :: "v"(lds_off), "v"(gaddr) : "memory");
}
__device__ __forceinline__ void wait_async0() {
    asm volatile("s_wait_asynccnt 0x0" ::: "memory");
}

// ---------------- Tensor Data Mover (TDM) 2-D tile load ----------------
#if __has_builtin(__builtin_amdgcn_tensor_load_to_lds)
#define HAVE_TDM 1
typedef unsigned int v4u __attribute__((ext_vector_type(4)));
typedef int          v8i __attribute__((ext_vector_type(8)));
typedef int          v4i __attribute__((ext_vector_type(4)));

// D# per ISA ch.8: group0 {count=1, lds_addr, global_addr(57b), type=2},
// group1 {data_size=2B, tensor dims, tile 32x(rows), dim0 stride}; groups 2/3
// zero (2-D tensor). Issued per-wave (EXEC ignored), tracked with TENSORcnt.
// This toolchain (clang-23 / therock-10.0) uses the 6-arg builtin form:
// (uint32x4 g0, int32x8 g1, int32x4, int32x4, int32x8, i32 cpol).
__device__ __forceinline__ void tdm_load_2d(unsigned lds_addr, const void* gaddr,
                                            unsigned tile_rows, unsigned tensor_d0,
                                            unsigned tensor_d1, unsigned stride_elems) {
    unsigned long long ga = (unsigned long long)gaddr;
    v4u g0;
    g0[0] = 1u;                                   // count=1, user mode, no gather
    g0[1] = lds_addr;                             // LDS byte address
    g0[2] = (unsigned)(ga & 0xffffffffu);         // global_addr[31:0]
    g0[3] = (unsigned)(ga >> 32) | (2u << 30);    // global_addr[56:32] | type=2
    v8i g1;
    g1[0] = (int)(1u << 16);                      // data_size=1 -> 2 bytes/elem
    g1[1] = (int)((tensor_d0 & 0xffffu) << 16);   // tensor_dim0[15:0]
    g1[2] = (int)((tensor_d0 >> 16) | ((tensor_d1 & 0xffffu) << 16));
    g1[3] = (int)((tensor_d1 >> 16) | (32u << 16)); // tile_dim0 = 32 elems
    g1[4] = (int)(tile_rows & 0xffffu);           // tile_dim1, tile_dim2=0
    g1[5] = (int)stride_elems;                    // tensor_dim0_stride[31:0]
    g1[6] = 0;
    g1[7] = 0;
    v4i z4 = {0, 0, 0, 0};
    v8i z8 = {0, 0, 0, 0, 0, 0, 0, 0};
    __builtin_amdgcn_tensor_load_to_lds(g0, g1, z4, z4, z8, 0);
}
__device__ __forceinline__ void wait_tensor0() {
    __builtin_amdgcn_s_wait_tensorcnt(0);
}
#else
#define HAVE_TDM 0
#endif

// Load a 16x32 bf16 WMMA operand fragment for this lane from a row-major
// [rows][rowStride] bf16 matrix (global or LDS). Per ISA 7.12.2: lane<16 holds
// K-chunks {0..7,16..23}; lane>=16 holds {8..15,24..31} of its row.
__device__ __forceinline__ Frag load_frag(const u16* base_row, int k, int half) {
    Frag f;
    f.q[0] = *reinterpret_cast<const uint4*>(base_row + k + half * 8);
    f.q[1] = *reinterpret_cast<const uint4*>(base_row + k + 16 + half * 8);
    return f;
}

// ---------------- one-time prep kernels ----------------
__global__ void k_cast_bf16(u16* __restrict__ dst, const float* __restrict__ src, int n) {
    for (int i = blockIdx.x * blockDim.x + threadIdx.x; i < n; i += gridDim.x * blockDim.x)
        dst[i] = f2bf(src[i]);
}

__global__ void k_w1pad(u16* __restrict__ dst, const float* __restrict__ W1) {
    const int n = H1d * KP1;
    for (int i = blockIdx.x * blockDim.x + threadIdx.x; i < n; i += gridDim.x * blockDim.x) {
        int r = i / KP1, c = i % KP1;
        dst[i] = (c < SIN) ? f2bf(W1[r * SIN + c]) : (u16)0;
    }
}

__global__ void k_aneg(float* __restrict__ Aneg, const float* __restrict__ A_log) {
    int i = blockIdx.x * blockDim.x + threadIdx.x;
    if (i < DD) Aneg[i] = -__expf(A_log[i]);
}

// x layout: [B][KP1] bf16. cols: 0..255 z | 256..511 z_static | 512..575 u | 576 dt | 577.. 0
__global__ void k_xstatic(u16* __restrict__ x, const float* __restrict__ zs,
                          const float* __restrict__ dt) {
    const int n = BB * (DS + 32);
    for (int i = blockIdx.x * blockDim.x + threadIdx.x; i < n; i += gridDim.x * blockDim.x) {
        int b = i / (DS + 32), r = i % (DS + 32);
        if (r < DS) {
            x[b * KP1 + 256 + r] = f2bf(zs[b * DS + r]);
        } else {
            int col = 576 + (r - DS);
            x[b * KP1 + col] = (col == 576) ? f2bf(dt[b]) : (u16)0;
        }
    }
}

// ---------------- per-step small kernels ----------------
__global__ void k_xdyn(u16* __restrict__ x, const float* __restrict__ zsrc,
                       const float* __restrict__ Ut, const float* __restrict__ dt,
                       float* __restrict__ utdt) {
    const int n = BB * (DD + DU);
    for (int i = blockIdx.x * blockDim.x + threadIdx.x; i < n; i += gridDim.x * blockDim.x) {
        int b = i / (DD + DU), r = i % (DD + DU);
        if (r < DD) {
            x[b * KP1 + r] = f2bf(zsrc[b * DD + r]);
        } else {
            int u = r - DD;
            float uv = Ut[b * DU + u];
            x[b * KP1 + 512 + u] = f2bf(uv);
            utdt[b * DU + u] = uv * dt[b];
        }
    }
}

__global__ void k_bnstats(const float* __restrict__ pre, float* __restrict__ mean,
                          float* __restrict__ var, int N) {
    __shared__ float ss[256], sq[256];
    int c = blockIdx.x, t = threadIdx.x;
    float s = 0.f, q = 0.f;
    for (int b = t; b < BB; b += 256) {
        float v = pre[b * N + c];
        s += v; q += v * v;
    }
    ss[t] = s; sq[t] = q;
    __syncthreads();
    for (int off = 128; off > 0; off >>= 1) {
        if (t < off) { ss[t] += ss[t + off]; sq[t] += sq[t + off]; }
        __syncthreads();
    }
    if (t == 0) {
        float m = ss[0] * (1.f / BB);
        mean[c] = m;
        var[c] = sq[0] * (1.f / BB) - m * m;
    }
}

__global__ void k_bnapply(u16* __restrict__ h, const float* __restrict__ pre,
                          const float* __restrict__ g, const float* __restrict__ be,
                          const float* __restrict__ mean, const float* __restrict__ var, int N) {
    const int n = BB * N;
    for (int i = blockIdx.x * blockDim.x + threadIdx.x; i < n; i += gridDim.x * blockDim.x) {
        int c = i % N;
        float v = g[c] * (pre[i] - mean[c]) * __frsqrt_rn(var[c] + EPSbn) + be[c];
        h[i] = f2bf(v > 0.f ? v : 0.f);
    }
}

__global__ void k_state(float* __restrict__ zout, const float* __restrict__ zprev,
                        const float* __restrict__ delta, const float* __restrict__ Aneg,
                        const float* __restrict__ zB) {
    const int n = BB * DD;
    for (int i = blockIdx.x * blockDim.x + threadIdx.x; i < n; i += gridDim.x * blockDim.x) {
        int d = i & (DD - 1);
        float dl = delta[i];
        zout[i] = __expf(dl * Aneg[d]) * zprev[i] + dl * zB[i];
    }
}

__global__ void k_yinit(float* __restrict__ Y, const float* __restrict__ utdt,
                        const float* __restrict__ Dm) {
    const int n = BB * NOBS;
    for (int i = blockIdx.x * blockDim.x + threadIdx.x; i < n; i += gridDim.x * blockDim.x) {
        int b = i / NOBS, o = i % NOBS;
        float s = 0.f;
        #pragma unroll 8
        for (int u = 0; u < DU; u++) s += utdt[b * DU + u] * Dm[o * DU + u];
        Y[i] = s;
    }
}

// ---------------- WMMA GEMM: out[M,N] = A[M,Kp] x W[N,Kp]^T + bias ----------------
// block = 256 threads (8 waves stacked on M), block tile 128(M) x 64(N).
// The 64x32 W tile shared by all 8 waves is DMA'd into LDS by the Tensor Data
// Mover (one descriptor per k-tile, wave 0 issues, TENSORcnt-tracked), double
// buffered; A rows loaded direct per wave.
template <int ACT, int OUTB>   // ACT: 0=none 1=softplus ; OUTB: 1 -> bf16 out
__global__ void k_gemm(const u16* __restrict__ A, const u16* __restrict__ W,
                       const float* __restrict__ bias, float* __restrict__ outF,
                       u16* __restrict__ outBf, int N, int Kp) {
    __shared__ u16 wtile[2][64 * 32];
    const int tid = threadIdx.x;
    const int L = tid & 31, w = tid >> 5;
    const int m0 = blockIdx.y * 128 + w * 16;
    const int n0 = blockIdx.x * 64;
    const int half = L >> 4, lr = L & 15;
    const int kt = Kp >> 5;

    const unsigned ldsA = (unsigned)(uintptr_t)&wtile[0][0];
    const unsigned ldsB = (unsigned)(uintptr_t)&wtile[1][0];
    const u16* wbase = W + (size_t)n0 * Kp;       // tile (t) starts at wbase + t*32
    const u16* arow = A + (size_t)(m0 + lr) * Kp;

    v8f acc[4] = {};
#if HAVE_TDM
    const bool issuer = (w == 0);
    if (issuer) tdm_load_2d(ldsA, wbase, 64, Kp, N - n0, Kp);
    for (int t = 0; t < kt; t++) {
        if (issuer) wait_tensor0();
        __syncthreads();                           // tile t resident for all waves
        if (issuer && t + 1 < kt)
            tdm_load_2d((t & 1) ? ldsA : ldsB, wbase + (t + 1) * 32, 64, Kp, N - n0, Kp);
#else
    const u16* wsrc = W + (size_t)(n0 + (tid >> 2)) * Kp + (tid & 3) * 8;
    const unsigned l0 = ldsA + tid * 16, l1 = ldsB + tid * 16;
    async_b128(l0, wsrc);
    for (int t = 0; t < kt; t++) {
        wait_async0();
        __syncthreads();
        if (t + 1 < kt) async_b128((t & 1) ? l0 : l1, wsrc + (t + 1) * 32);
#endif
        __builtin_prefetch(arow + t * 32 + 128, 0, 3);
        const u16* wt = &wtile[t & 1][0];
        Frag a  = load_frag(arow, t * 32, half);
        Frag b0 = load_frag(wt + (0 * 16 + lr) * 32, 0, half);
        Frag b1 = load_frag(wt + (1 * 16 + lr) * 32, 0, half);
        Frag b2 = load_frag(wt + (2 * 16 + lr) * 32, 0, half);
        Frag b3 = load_frag(wt + (3 * 16 + lr) * 32, 0, half);
        acc[0] = __builtin_amdgcn_wmma_f32_16x16x32_bf16(false, a.v, false, b0.v, (short)0, acc[0], false, false);
        acc[1] = __builtin_amdgcn_wmma_f32_16x16x32_bf16(false, a.v, false, b1.v, (short)0, acc[1], false, false);
        acc[2] = __builtin_amdgcn_wmma_f32_16x16x32_bf16(false, a.v, false, b2.v, (short)0, acc[2], false, false);
        acc[3] = __builtin_amdgcn_wmma_f32_16x16x32_bf16(false, a.v, false, b3.v, (short)0, acc[3], false, false);
        __syncthreads();                           // protect buffer before re-issue
    }

    const int mh = half * 8;
    #pragma unroll
    for (int j = 0; j < 4; j++) {
        int ncol = n0 + j * 16 + lr;
        float bia = bias[ncol];
        #pragma unroll
        for (int i = 0; i < 8; i++) {
            int row = m0 + mh + i;
            float v = acc[j][i] + bia;
            if (ACT == 1) v = (v > 20.f) ? v : log1pf(__expf(v));
            if (OUTB) outBf[(size_t)row * N + ncol] = f2bf(v);
            else      outF [(size_t)row * N + ncol] = v;
        }
    }
}

// ---------------- fused Wb GEMM + einsum('bdu,bu->bd') ----------------
// wave tile: 16 b-rows x 64 u-cols for a single d; zB[b,d] = sum_u (hz.Wb^T + bb)*utdt
__global__ void k_gemm_bu(const u16* __restrict__ hz, const u16* __restrict__ Wb,
                          const float* __restrict__ bb, const float* __restrict__ utdt,
                          float* __restrict__ zB) {
    __shared__ u16 atile[2][16 * 32];
    const int tid = threadIdx.x;
    const int L = tid & 31, w = tid >> 5;
    const int d = (blockIdx.x * 8 + w) & (DD - 1);
    const int m0 = (blockIdx.x >> 5) * 16;        // b-block, same for all 8 waves
    const int n0 = d * DU;
    const int half = L >> 4, lr = L & 15;

    const u16* asrc = hz + (size_t)(m0 + (tid >> 2)) * HZd + (tid & 3) * 8;
    const unsigned lds0 = (unsigned)(uintptr_t)&atile[0][0] + tid * 16;
    const unsigned lds1 = (unsigned)(uintptr_t)&atile[1][0] + tid * 16;

    const u16* wrow[4];
    #pragma unroll
    for (int j = 0; j < 4; j++) wrow[j] = Wb + (size_t)(n0 + j * 16 + lr) * HZd;

    v8f acc[4] = {};
    if (tid < 64) async_b128(lds0, asrc);
    for (int t = 0; t < (HZd >> 5); t++) {
        wait_async0();
        __syncthreads();
        if (tid < 64 && t + 1 < (HZd >> 5))
            async_b128((t & 1) ? lds0 : lds1, asrc + (t + 1) * 32);
        int k = t * 32;
        __builtin_prefetch(wrow[0] + k + 128, 0, 3);
        Frag a  = load_frag(&atile[t & 1][lr * 32], 0, half);
        Frag b0 = load_frag(wrow[0], k, half);
        Frag b1 = load_frag(wrow[1], k, half);
        Frag b2 = load_frag(wrow[2], k, half);
        Frag b3 = load_frag(wrow[3], k, half);
        acc[0] = __builtin_amdgcn_wmma_f32_16x16x32_bf16(false, a.v, false, b0.v, (short)0, acc[0], false, false);
        acc[1] = __builtin_amdgcn_wmma_f32_16x16x32_bf16(false, a.v, false, b1.v, (short)0, acc[1], false, false);
        acc[2] = __builtin_amdgcn_wmma_f32_16x16x32_bf16(false, a.v, false, b2.v, (short)0, acc[2], false, false);
        acc[3] = __builtin_amdgcn_wmma_f32_16x16x32_bf16(false, a.v, false, b3.v, (short)0, acc[3], false, false);
        __syncthreads();
    }

    const int mh = half * 8;
    float s[8] = {0.f, 0.f, 0.f, 0.f, 0.f, 0.f, 0.f, 0.f};
    #pragma unroll
    for (int j = 0; j < 4; j++) {
        int u = j * 16 + lr;
        float bbv = bb[d * DU + u];
        #pragma unroll
        for (int i = 0; i < 8; i++) {
            int b = m0 + mh + i;
            s[i] += (acc[j][i] + bbv) * utdt[b * DU + u];
        }
    }
    #pragma unroll
    for (int m = 1; m <= 8; m <<= 1)
        #pragma unroll
        for (int i = 0; i < 8; i++) s[i] += __shfl_xor(s[i], m, 32);
    if (lr == 0)
        #pragma unroll
        for (int i = 0; i < 8; i++) zB[(size_t)(m0 + mh + i) * DD + d] = s[i];
}

// ---------------- fused Wc GEMM + einsum('bod,bd->bo') ----------------
__global__ void k_gemm_cy(const u16* __restrict__ hz, const u16* __restrict__ Wc,
                          const float* __restrict__ bc, const float* __restrict__ zn,
                          float* __restrict__ Y) {
    __shared__ u16 atile[2][16 * 32];
    const int tid = threadIdx.x;
    const int L = tid & 31, w = tid >> 5;
    const int base = blockIdx.x * 8;
    const int m0 = (base / (NOBS * 4)) * 16;      // same for all 8 waves (8 | 320)
    const int rem = base - (base / (NOBS * 4)) * (NOBS * 4) + w;
    const int o = rem >> 2, q = rem & 3;
    const int n0 = o * DD + q * 64;
    const int half = L >> 4, lr = L & 15;

    const u16* asrc = hz + (size_t)(m0 + (tid >> 2)) * HZd + (tid & 3) * 8;
    const unsigned lds0 = (unsigned)(uintptr_t)&atile[0][0] + tid * 16;
    const unsigned lds1 = (unsigned)(uintptr_t)&atile[1][0] + tid * 16;

    const u16* wrow[4];
    #pragma unroll
    for (int j = 0; j < 4; j++) wrow[j] = Wc + (size_t)(n0 + j * 16 + lr) * HZd;

    v8f acc[4] = {};
    if (tid < 64) async_b128(lds0, asrc);
    for (int t = 0; t < (HZd >> 5); t++) {
        wait_async0();
        __syncthreads();
        if (tid < 64 && t + 1 < (HZd >> 5))
            async_b128((t & 1) ? lds0 : lds1, asrc + (t + 1) * 32);
        int k = t * 32;
        __builtin_prefetch(wrow[0] + k + 128, 0, 3);
        Frag a  = load_frag(&atile[t & 1][lr * 32], 0, half);
        Frag b0 = load_frag(wrow[0], k, half);
        Frag b1 = load_frag(wrow[1], k, half);
        Frag b2 = load_frag(wrow[2], k, half);
        Frag b3 = load_frag(wrow[3], k, half);
        acc[0] = __builtin_amdgcn_wmma_f32_16x16x32_bf16(false, a.v, false, b0.v, (short)0, acc[0], false, false);
        acc[1] = __builtin_amdgcn_wmma_f32_16x16x32_bf16(false, a.v, false, b1.v, (short)0, acc[1], false, false);
        acc[2] = __builtin_amdgcn_wmma_f32_16x16x32_bf16(false, a.v, false, b2.v, (short)0, acc[2], false, false);
        acc[3] = __builtin_amdgcn_wmma_f32_16x16x32_bf16(false, a.v, false, b3.v, (short)0, acc[3], false, false);
        __syncthreads();
    }

    const int mh = half * 8;
    float s[8] = {0.f, 0.f, 0.f, 0.f, 0.f, 0.f, 0.f, 0.f};
    #pragma unroll
    for (int j = 0; j < 4; j++) {
        int dcol = q * 64 + j * 16 + lr;
        float bcv = bc[o * DD + dcol];
        #pragma unroll
        for (int i = 0; i < 8; i++) {
            int b = m0 + mh + i;
            s[i] += (acc[j][i] + bcv) * zn[(size_t)b * DD + dcol];
        }
    }
    #pragma unroll
    for (int m = 1; m <= 8; m <<= 1)
        #pragma unroll
        for (int i = 0; i < 8; i++) s[i] += __shfl_xor(s[i], m, 32);
    if (lr == 0)
        #pragma unroll
        for (int i = 0; i < 8; i++)
            atomicAdd(&Y[(size_t)(m0 + mh + i) * NOBS + o], s[i]);
}

// ---------------- host launch ----------------
extern "C" void kernel_launch(void* const* d_in, const int* in_sizes, int n_in,
                              void* d_out, int out_size, void* d_ws, size_t ws_size,
                              hipStream_t stream) {
    const float* z_dyn = (const float*)d_in[0];
    const float* z_st  = (const float*)d_in[1];
    const float* dt    = (const float*)d_in[2];
    const float* U     = (const float*)d_in[3];
    const float* W1    = (const float*)d_in[4];
    const float* b1    = (const float*)d_in[5];
    const float* g1    = (const float*)d_in[6];
    const float* be1   = (const float*)d_in[7];
    const float* W2    = (const float*)d_in[8];
    const float* b2    = (const float*)d_in[9];
    const float* g2    = (const float*)d_in[10];
    const float* be2   = (const float*)d_in[11];
    const float* W3    = (const float*)d_in[12];
    const float* b3    = (const float*)d_in[13];
    const float* A_log = (const float*)d_in[14];
    const float* Wd    = (const float*)d_in[15];
    const float* bd    = (const float*)d_in[16];
    const float* Wb    = (const float*)d_in[17];
    const float* bb    = (const float*)d_in[18];
    const float* Wc    = (const float*)d_in[19];
    const float* bc    = (const float*)d_in[20];
    const float* Dm    = (const float*)d_in[21];

    float* Zs = (float*)d_out;                        // [T,B,DD]
    float* Ys = (float*)d_out + (size_t)TT * BB * DD; // [T,B,NOBS]

    char* p = (char*)d_ws;
    size_t off = 0;
    auto carve = [&](size_t bytes) {
        void* r = p + off;
        off += (bytes + 255) & ~(size_t)255;
        return r;
    };
    u16*   W1b  = (u16*)carve((size_t)H1d * KP1 * 2);
    u16*   W2b  = (u16*)carve((size_t)H1d * H1d * 2);
    u16*   W3b  = (u16*)carve((size_t)HZd * H1d * 2);
    u16*   Wdb  = (u16*)carve((size_t)DD * HZd * 2);
    u16*   Wbb  = (u16*)carve((size_t)NB_ * HZd * 2);
    u16*   Wcb  = (u16*)carve((size_t)NC_ * HZd * 2);
    float* Aneg = (float*)carve(DD * 4);
    u16*   xb   = (u16*)carve((size_t)BB * KP1 * 2);
    float* pre  = (float*)carve((size_t)BB * H1d * 4);
    float* mean = (float*)carve(H1d * 4);
    float* var  = (float*)carve(H1d * 4);
    u16*   hb   = (u16*)carve((size_t)BB * H1d * 2);
    u16*   hzb  = (u16*)carve((size_t)BB * HZd * 2);
    float* delta= (float*)carve((size_t)BB * DD * 4);
    float* utdt = (float*)carve((size_t)BB * DU * 4);
    float* zB   = (float*)carve((size_t)BB * DD * 4);
    (void)ws_size; (void)in_sizes; (void)n_in; (void)out_size;

    // ---- one-time prep ----
    k_w1pad   <<<2048, 256, 0, stream>>>(W1b, W1);
    k_cast_bf16<<<2048, 256, 0, stream>>>(W2b, W2, H1d * H1d);
    k_cast_bf16<<<2048, 256, 0, stream>>>(W3b, W3, HZd * H1d);
    k_cast_bf16<<<2048, 256, 0, stream>>>(Wdb, Wd, DD * HZd);
    k_cast_bf16<<<4096, 256, 0, stream>>>(Wbb, Wb, NB_ * HZd);
    k_cast_bf16<<<4096, 256, 0, stream>>>(Wcb, Wc, NC_ * HZd);
    k_aneg    <<<1, 256, 0, stream>>>(Aneg, A_log);
    k_xstatic <<<1024, 256, 0, stream>>>(xb, z_st, dt);

    // ---- sequential scan over T steps ----
    for (int t = 0; t < TT; t++) {
        const float* zprev = (t == 0) ? z_dyn : (Zs + (size_t)(t - 1) * BB * DD);
        const float* Ut = U + (size_t)t * BB * DU;
        float* zout = Zs + (size_t)t * BB * DD;
        float* yout = Ys + (size_t)t * BB * NOBS;

        k_xdyn<<<1024, 256, 0, stream>>>(xb, zprev, Ut, dt, utdt);

        // layer 1: pre = x @ W1^T + b1 ; BN ; ReLU
        k_gemm<0, 0><<<dim3(H1d / 64, BB / 128), 256, 0, stream>>>(xb, W1b, b1, pre, nullptr, H1d, KP1);
        k_bnstats<<<H1d, 256, 0, stream>>>(pre, mean, var, H1d);
        k_bnapply<<<1024, 256, 0, stream>>>(hb, pre, g1, be1, mean, var, H1d);

        // layer 2
        k_gemm<0, 0><<<dim3(H1d / 64, BB / 128), 256, 0, stream>>>(hb, W2b, b2, pre, nullptr, H1d, H1d);
        k_bnstats<<<H1d, 256, 0, stream>>>(pre, mean, var, H1d);
        k_bnapply<<<1024, 256, 0, stream>>>(hb, pre, g2, be2, mean, var, H1d);

        // hz (bf16 out, feeds 3 GEMMs)
        k_gemm<0, 1><<<dim3(HZd / 64, BB / 128), 256, 0, stream>>>(hb, W3b, b3, nullptr, hzb, HZd, H1d);

        // delta = softplus(hz @ Wd^T + bd)
        k_gemm<1, 0><<<dim3(DD / 64, BB / 128), 256, 0, stream>>>(hzb, Wdb, bd, delta, nullptr, DD, HZd);

        // zB[b,d] = sum_u (hz@Wb^T + bb)[b,d,u] * utdt[b,u]   (fused, Bt never materialized)
        k_gemm_bu<<<(BB / 16) * DD / 8, 256, 0, stream>>>(hzb, Wbb, bb, utdt, zB);

        // z_next = exp(delta*A)*z + delta*zB  -> Zs[t]
        k_state<<<1024, 256, 0, stream>>>(zout, zprev, delta, Aneg, zB);

        // y = utdt @ D^T, then += sum_d (hz@Wc^T + bc)[b,o,d] * z_next[b,d]
        k_yinit<<<320, 256, 0, stream>>>(yout, utdt, Dm);
        k_gemm_cy<<<(BB / 16) * NOBS * 4 / 8, 256, 0, stream>>>(hzb, Wcb, bc, zout, yout);
    }
}
